// DotProductAttention_36155034698064
// MI455X (gfx1250) — matrix operand order repaired and checked by reading, not compiled, not run on hardware
//
#include <hip/hip_runtime.h>

typedef __attribute__((ext_vector_type(16))) _Float16 v16h;
typedef __attribute__((ext_vector_type(8)))  float    v8f;

#define B_N 32
#define S_N 2048
#define D_N 64
#define INV_SCALE 0.125f

__device__ __forceinline__ v8f wmma_f16(v16h a, v16h b, v8f c) {
    // v_wmma_f32_16x16x32_f16: D = A(16x32 f16) * B(32x16 f16) + C(16x16 f32)
    return __builtin_amdgcn_wmma_f32_16x16x32_f16(
        /*neg_a=*/false, a, /*neg_b=*/false, b,
        /*c_mod=*/(short)0, c, /*reuse_a=*/false, /*reuse_b=*/false);
}

__device__ __forceinline__ void cvt8(v16h& d, int base, float4 u, float4 v) {
    d[base + 0] = (_Float16)u.x; d[base + 1] = (_Float16)u.y;
    d[base + 2] = (_Float16)u.z; d[base + 3] = (_Float16)u.w;
    d[base + 4] = (_Float16)v.x; d[base + 5] = (_Float16)v.y;
    d[base + 6] = (_Float16)v.z; d[base + 7] = (_Float16)v.w;
}

// 16 contiguous f32 -> 16 packed f16 (B-matrix fragment: element t = K index t)
__device__ __forceinline__ v16h bfrag16(const float* __restrict__ p) {
    const float4* p4 = (const float4*)p;
    float4 a = p4[0], b = p4[1], c = p4[2], d = p4[3];
    v16h r;
    cvt8(r, 0, a, b);
    cvt8(r, 8, c, d);
    return r;
}

// V fragment: column walk with stride D (element t = key t of this 16-key half)
__device__ __forceinline__ v16h vfrag(const float* __restrict__ p) {
    v16h r;
#pragma unroll
    for (int t = 0; t < 16; ++t) r[t] = (_Float16)p[t * D_N];
    return r;
}

__device__ __forceinline__ float redmax16(float x) {
#pragma unroll
    for (int m = 1; m < 16; m <<= 1) x = fmaxf(x, __shfl_xor(x, m, 32));
    return x;
}
__device__ __forceinline__ float redsum16(float x) {
#pragma unroll
    for (int m = 1; m < 16; m <<= 1) x += __shfl_xor(x, m, 32);
    return x;
}

__global__ __launch_bounds__(256)
void DotProductAttention_kernel(const float* __restrict__ Q,
                                const float* __restrict__ K,
                                const float* __restrict__ V,
                                float* __restrict__ out,
                                float* __restrict__ attn) {
    __shared__ _Float16 ldsP[8][16 * 32];  // per-wave P transpose scratch (1 KB each)

    const int wave = threadIdx.x >> 5;
    const int lane = threadIdx.x & 31;
    const int l15  = lane & 15;
    const int hi   = lane >> 4;          // 0: rows 0-7 / K 0-15 half, 1: rows 8-15 / K 16-31 half
    const int qt   = blockIdx.x * 8 + wave;  // 16-row query tile index
    const int b    = blockIdx.y;
    const int q0   = qt << 4;

    const float* Qb = Q + ((size_t)b * S_N + q0) * D_N;
    const float* Kb = K + (size_t)b * S_N * D_N;
    const float* Vb = V + (size_t)b * S_N * D_N;
    float* outb  = out  + ((size_t)b * S_N + q0) * D_N;
    float* attnb = attn + ((size_t)b * S_N + q0) * (size_t)S_N;

    // ---- Q A-fragments (16x32 f16 each, contiguous k order) -----------------------
    // lane<16 : t=0..15 -> K=t
    // lane>=16: t=0..15 -> K=16+t
    const float* qrow = Qb + l15 * D_N;
    v16h a0, a1;
    {
        const float4* g0 = (const float4*)(qrow + hi * 16);        // K base 0
        const float4* g1 = (const float4*)(qrow + 8 + hi * 16);
        const float4* g2 = (const float4*)(qrow + 32 + hi * 16);   // K base 32
        const float4* g3 = (const float4*)(qrow + 40 + hi * 16);
        cvt8(a0, 0, g0[0], g0[1]);
        cvt8(a0, 8, g1[0], g1[1]);
        cvt8(a1, 0, g2[0], g2[1]);
        cvt8(a1, 8, g3[0], g3[1]);
    }

    v8f acc0 = {}, acc1 = {}, acc2 = {}, acc3 = {};   // out accum, D chunks 0..3
    float m8[8], l8[8];
#pragma unroll
    for (int r = 0; r < 8; ++r) { m8[r] = -3.0e38f; l8[r] = 0.0f; }

    const int n32  = (qt + 2) >> 1;       // # of 32-key tiles needed (causal)
    const int rowq = q0 + hi * 8;         // + r = this lane's C-layout row

    // ======================= Pass 1: flash attention =========================
    for (int kt = 0; kt < n32; ++kt) {
        const int kb = kt * 32;
        const float* krow0 = Kb + (size_t)(kb + l15) * D_N + hi * 16;
        const float* krow1 = krow0 + 16 * D_N;
        if (kt + 1 < n32)
            __builtin_prefetch((const void*)(krow0 + 32 * D_N), 0, 3);

        // K B-fragments: B(k=dim, n=key)
        v16h b00 = bfrag16(krow0),      b01 = bfrag16(krow0 + 32);
        v16h b10 = bfrag16(krow1),      b11 = bfrag16(krow1 + 32);

        v8f cl = {}, ch = {};
        cl = wmma_f16(a0, b00, cl); cl = wmma_f16(a1, b01, cl);   // keys kb..kb+15
        ch = wmma_f16(a0, b10, ch); ch = wmma_f16(a1, b11, ch);   // keys kb+16..kb+31

        const int col0 = kb + l15, col1 = col0 + 16;
        float mt[8];
#pragma unroll
        for (int r = 0; r < 8; ++r) {
            const int row = rowq + r;
            float sl = (col0 <= row) ? cl[r] * INV_SCALE : -__builtin_inff();
            float sh = (col1 <= row) ? ch[r] * INV_SCALE : -__builtin_inff();
            cl[r] = sl; ch[r] = sh;
            mt[r] = fmaxf(sl, sh);
        }
#pragma unroll
        for (int r = 0; r < 8; ++r) mt[r] = redmax16(mt[r]);

        float f8[8];
#pragma unroll
        for (int r = 0; r < 8; ++r) {
            float mn = fmaxf(m8[r], mt[r]);
            f8[r] = __expf(m8[r] - mn);
            m8[r] = mn;
        }
        float s8[8];
#pragma unroll
        for (int r = 0; r < 8; ++r) {
            float pl = __expf(cl[r] - m8[r]);
            float ph = __expf(ch[r] - m8[r]);
            cl[r] = pl; ch[r] = ph;
            s8[r] = pl + ph;
        }
#pragma unroll
        for (int r = 0; r < 8; ++r) s8[r] = redsum16(s8[r]);
#pragma unroll
        for (int r = 0; r < 8; ++r) {
            l8[r] = l8[r] * f8[r] + s8[r];
            acc0[r] *= f8[r]; acc1[r] *= f8[r];
            acc2[r] *= f8[r]; acc3[r] *= f8[r];
        }

        // ---- C-layout P -> A-layout P via per-wave LDS (16x32 f16) ----------
        _Float16* lp = ldsP[wave];
#pragma unroll
        for (int r = 0; r < 8; ++r) {
            lp[(hi * 8 + r) * 32 + l15]      = (_Float16)cl[r];
            lp[(hi * 8 + r) * 32 + 16 + l15] = (_Float16)ch[r];
        }
        __asm__ volatile("s_wait_dscnt 0" ::: "memory");
        v16h pa;
        {
            const _Float16* pr = lp + l15 * 32;
#pragma unroll
            for (int t = 0; t < 8; ++t) {
                pa[t]     = pr[hi * 16 + t];
                pa[8 + t] = pr[hi * 16 + 8 + t];
            }
        }

        // ---- P(16x32) x V(32x64): 4 WMMAs over D chunks ---------------------
        const float* vb0 = Vb + (size_t)(kb + hi * 16) * D_N + l15;
        acc0 = wmma_f16(pa, vfrag(vb0),      acc0);
        acc1 = wmma_f16(pa, vfrag(vb0 + 16), acc1);
        acc2 = wmma_f16(pa, vfrag(vb0 + 32), acc2);
        acc3 = wmma_f16(pa, vfrag(vb0 + 48), acc3);
    }

    // ---- normalize + write `out` -------------------------------------------
    float inv8[8];
#pragma unroll
    for (int r = 0; r < 8; ++r) inv8[r] = 1.0f / l8[r];
#pragma unroll
    for (int r = 0; r < 8; ++r) {
        const int row = hi * 8 + r;
        outb[row * D_N + l15]      = acc0[r] * inv8[r];
        outb[row * D_N + 16 + l15] = acc1[r] * inv8[r];
        outb[row * D_N + 32 + l15] = acc2[r] * inv8[r];
        outb[row * D_N + 48 + l15] = acc3[r] * inv8[r];
    }

    // ============ Pass 2: recompute scores, emit attn probabilities ==========
    for (int kt = 0; kt < n32; ++kt) {
        const int kb = kt * 32;
        const float* krow0 = Kb + (size_t)(kb + l15) * D_N + hi * 16;
        const float* krow1 = krow0 + 16 * D_N;
        v16h b00 = bfrag16(krow0),      b01 = bfrag16(krow0 + 32);
        v16h b10 = bfrag16(krow1),      b11 = bfrag16(krow1 + 32);
        v8f cl = {}, ch = {};
        cl = wmma_f16(a0, b00, cl); cl = wmma_f16(a1, b01, cl);
        ch = wmma_f16(a0, b10, ch); ch = wmma_f16(a1, b11, ch);
        const int col0 = kb + l15, col1 = col0 + 16;
#pragma unroll
        for (int r = 0; r < 8; ++r) {
            const int row = rowq + r;
            float pl = (col0 <= row) ? __expf(cl[r] * INV_SCALE - m8[r]) * inv8[r] : 0.0f;
            float ph = (col1 <= row) ? __expf(ch[r] * INV_SCALE - m8[r]) * inv8[r] : 0.0f;
            float* arow = attnb + (size_t)(hi * 8 + r) * S_N;
            arow[col0] = pl;
            arow[col1] = ph;
        }
    }

    // ---- zero-fill strictly-masked rectangle (cols >= n32*32) ---------------
    const int c0 = n32 * 32;
    const float4 z4 = make_float4(0.0f, 0.0f, 0.0f, 0.0f);
    for (int row = 0; row < 16; ++row) {
        float* rp = attnb + (size_t)row * S_N;
        for (int c = c0 + lane * 4; c < S_N; c += 128)
            *(float4*)(rp + c) = z4;
    }
}

extern "C" void kernel_launch(void* const* d_in, const int* in_sizes, int n_in,
                              void* d_out, int out_size, void* d_ws, size_t ws_size,
                              hipStream_t stream) {
    const float* q = (const float*)d_in[0];
    const float* k = (const float*)d_in[1];
    const float* v = (const float*)d_in[2];
    // d_in[3] is the causal mask; causality is computed analytically in-kernel.
    float* out  = (float*)d_out;
    float* attn = out + (size_t)B_N * S_N * D_N;  // tuple: (out, attn) concatenated

    dim3 grid(S_N / (16 * 8), B_N);  // 8 waves/block, one 16-row q-tile per wave
    DotProductAttention_kernel<<<grid, 256, 0, stream>>>(q, k, v, out, attn);
}
